// FiniteWindowTopologicalAttention_7687991459969
// MI455X (gfx1250) — matrix operand order repaired and checked by reading, not compiled, not run on hardware
//
#include <hip/hip_runtime.h>
#include <hip/hip_bf16.h>
#include <cstdint>

// ---------- WMMA types (CDNA5 / gfx1250, wave32) ----------
typedef __attribute__((ext_vector_type(16))) __bf16 v16bf;
typedef __attribute__((ext_vector_type(8)))  float  v8f;

#define WMMA_BF16(a, b, c) \
    __builtin_amdgcn_wmma_f32_16x16x32_bf16(false, (a), false, (b), (short)0, (c), false, false)

// Problem constants
#define BATCH 4
#define TDIM 256
#define SDIM 256
#define D 256
#define NH 8
#define HD 32
#define W 16                    // tokens per 4x4 window
#define NWIN_TOTAL 16384        // 4 * 64 * 64
#define WPB 4                   // windows per block (batched along window-column)
#define MROWS 64                // WPB * W rows per block
#define NBLK (NWIN_TOTAL / WPB) // 4096
#define QKV_N 768
#define KCH 8                   // 256/32 K-chunks

// LDS strides (bf16 elems; rows 16B-aligned)
#define XY_STRIDE 264           // shared x-tile / y-tile buffer
#define QK_STRIDE 520           // q,k only (512 + pad)
#define VT_STRIDE 16            // vT row: 16 keys

// ---------- fragment loaders (contiguous k order, same for A and B) ----------
// A 16x32 bf16: lanes<16 K{0..15}, lanes>=16 K{16..31}; p pre-offset by hi?16:0.
static __device__ __forceinline__ v16bf frag_a(const __bf16* p) {
    union { uint4 u[2]; v16bf v; } f;
    f.u[0] = *(const uint4*)(p);
    f.u[1] = *(const uint4*)(p + 8);
    return f.v;
}
// B 32x16 bf16: per lane 16 contiguous K elems (p pre-offset by hi?16:0 when from row-major).
static __device__ __forceinline__ v16bf frag_b(const __bf16* p) {
    union { uint4 u[2]; v16bf v; } f;
    f.u[0] = *(const uint4*)(p);
    f.u[1] = *(const uint4*)(p + 8);
    return f.v;
}
// low 8 K-elems, upper half zero (K padding)
static __device__ __forceinline__ v16bf frag_lo(const __bf16* p) {
    union { uint4 u[2]; v16bf v; } f;
    f.u[0] = *(const uint4*)(p);
    f.u[1] = make_uint4(0u, 0u, 0u, 0u);
    return f.v;
}
static __device__ __forceinline__ v16bf frag_zero() {
    union { uint4 u[2]; v16bf v; } f;
    f.u[0] = make_uint4(0u, 0u, 0u, 0u);
    f.u[1] = f.u[0];
    return f.v;
}

// ---------- prep: pack fp32 weights -> bf16 per-lane B-fragment order ----------
// dst[((ntile*8+kc)*32+lane)*16+e] = src[K*ncols+N],
// K = kc*32 + (lane<16?0:16) + e ; N = ntile*16 + (lane&15)
__global__ void pack_weights_kernel(const float* __restrict__ src, __bf16* __restrict__ dst,
                                    int ntiles, int ncols) {
    int idx = blockIdx.x * 256 + threadIdx.x;
    int total = ntiles * KCH * 32 * 16;
    if (idx >= total) return;
    int e     = idx & 15;
    int lane  = (idx >> 4) & 31;
    int kc    = (idx >> 9) & 7;
    int ntile = idx >> 12;
    int K = kc * 32 + ((lane < 16) ? 0 : 16) + e;
    int N = ntile * 16 + (lane & 15);
    dst[idx] = (__bf16)src[K * ncols + N];
}

// ---------- fused kernel: one workgroup per 4 windows (M=64), 16 waves ----------
__global__ __launch_bounds__(512)
void window_attn_kernel(const float* __restrict__ x,
                        const __bf16* __restrict__ wqkv_pk,   // packed 48x8x32x16
                        const __bf16* __restrict__ wproj_pk,  // packed 16x8x32x16
                        const float* __restrict__ b_proj,
                        const float* __restrict__ rpb_table,  // 49 x 8
                        float* __restrict__ out) {
    __shared__ float  rpbS[49 * NH];                 // 1.5 KB
    __shared__ __bf16 xyS[MROWS * XY_STRIDE];        // 33 KB : x-tile, later y-tile
    __shared__ __bf16 qkS[MROWS * QK_STRIDE];        // 65 KB : q (0..255), k (256..511)
    __shared__ __bf16 vT[WPB * D * VT_STRIDE];       // 32 KB : [win][channel][key]
    __shared__ __bf16 pS[WPB * NH * W * 16];         // 16 KB : softmax probs per pair

    const int tid  = threadIdx.x;
    const int lane = tid & 31;
    const int wave = tid >> 5;                       // 0..15
    const int m16  = lane & 15;
    const bool hi  = (lane >= 16);
    const int aoff = hi ? 16 : 0;

    // block -> 4 consecutive windows along window-column
    const int w0  = blockIdx.x * WPB;
    const int b   = w0 >> 12;
    const int wt  = (w0 >> 6) & 63;
    const int wc0 = w0 & 63;

    for (int i = tid; i < 49 * NH; i += 512) rpbS[i] = rpb_table[i];

    // ---- Phase A: load 64 tokens x 256 ch -> bf16 LDS ----
    {
        const int m  = tid >> 3;                     // row 0..63
        const int cg = (tid & 7) * 32;               // 32 channels / thread
        const int wi = m >> 4, p = m & 15;
        const int tp = p >> 2, sp = p & 3;
        const size_t row = (((size_t)b * TDIM) + (wt * 4 + tp)) * SDIM + ((wc0 + wi) * 4 + sp);
        const float* src = x + row * D + cg;
        #pragma unroll
        for (int v = 0; v < 8; ++v) {
            float4 f = *(const float4*)(src + v * 4);
            __bf16* dst = &xyS[m * XY_STRIDE + cg + v * 4];
            dst[0] = (__bf16)f.x; dst[1] = (__bf16)f.y;
            dst[2] = (__bf16)f.z; dst[3] = (__bf16)f.w;
        }
    }
    __syncthreads();

    // ---- Phase B: QKV GEMM (64x256)@(256x768); each B-frag reused for 4 M-tiles ----
    #pragma unroll
    for (int t = 0; t < 3; ++t) {
        const int ntile = wave * 3 + t;              // 0..47
        v8f acc[WPB] = {};
        #pragma unroll
        for (int kc = 0; kc < KCH; ++kc) {
            const __bf16* bp = &wqkv_pk[(size_t)((ntile * KCH + kc) * 32 + lane) * 16];
            if (kc < KCH - 1) __builtin_prefetch(bp + 512, 0, 0);
            v16bf bfr = frag_b(bp);
            #pragma unroll
            for (int mt = 0; mt < WPB; ++mt) {
                v16bf a = frag_a(&xyS[(mt * 16 + m16) * XY_STRIDE + kc * 32 + aoff]);
                acc[mt] = WMMA_BF16(a, bfr, acc[mt]);
            }
        }
        if (ntile < 32) {                            // q / k -> qkS
            const int n = ntile * 16 + m16;
            #pragma unroll
            for (int mt = 0; mt < WPB; ++mt)
                #pragma unroll
                for (int r = 0; r < 8; ++r) {
                    const int mr = mt * 16 + (hi ? 8 + r : r);
                    qkS[mr * QK_STRIDE + n] = (__bf16)acc[mt][r];
                }
        } else {                                     // v -> transposed vT[win][ch][key]
            const int ch = (ntile - 32) * 16 + m16;  // 0..255
            #pragma unroll
            for (int mt = 0; mt < WPB; ++mt)
                #pragma unroll
                for (int r = 0; r < 8; ++r) {
                    const int key = hi ? 8 + r : r;
                    vT[(mt * D + ch) * VT_STRIDE + key] = (__bf16)acc[mt][r];
                }
        }
    }
    __syncthreads();

    // ---- Phase C: attention; each wave handles 2 (window, head) pairs ----
    #pragma unroll
    for (int i = 0; i < 2; ++i) {
        const int pair = wave * 2 + i;               // 0..31
        const int wi = pair >> 3, h = pair & 7;
        const int rbase = wi * 16;

        // S = q @ k^T  (M=16, N=16 keys, K=hd=32)
        v16bf aq = frag_a(&qkS[(rbase + m16) * QK_STRIDE + h * HD + aoff]);
        v16bf bk = frag_b(&qkS[(rbase + m16) * QK_STRIDE + 256 + h * HD + (hi ? 16 : 0)]);
        v8f s = WMMA_BF16(aq, bk, (v8f){});

        const float scale = 0.17677669529663687f;    // 1/sqrt(32)
        const int nk = m16;
        #pragma unroll
        for (int r = 0; r < 8; ++r) {
            const int mq = hi ? 8 + r : r;
            const int rpi = ((mq >> 2) - (nk >> 2) + 3) * 7 + ((mq & 3) - (nk & 3) + 3);
            float val = s[r] * scale + rpbS[rpi * NH + h];
            float mx = val;
            #pragma unroll
            for (int o = 1; o < 16; o <<= 1) mx = fmaxf(mx, __shfl_xor(mx, o, 16));
            float ex = __expf(val - mx);
            float sm = ex;
            #pragma unroll
            for (int o = 1; o < 16; o <<= 1) sm += __shfl_xor(sm, o, 16);
            pS[pair * 256 + mq * 16 + nk] = (__bf16)(ex / sm);
        }

        // out = P(16x16, K-padded) @ V(16x32): two N=16 WMMAs
        v16bf ap = hi ? frag_zero() : frag_b(&pS[pair * 256 + m16 * 16]);
        #pragma unroll
        for (int chunk = 0; chunk < 2; ++chunk) {
            const int ch = h * HD + chunk * 16 + m16;
            v16bf bv = hi ? frag_zero() : frag_b(&vT[(wi * D + ch) * VT_STRIDE]);
            v8f o = WMMA_BF16(ap, bv, (v8f){});
            #pragma unroll
            for (int r = 0; r < 8; ++r) {
                const int mr = rbase + (hi ? 8 + r : r);
                xyS[mr * XY_STRIDE + h * HD + chunk * 16 + m16] = (__bf16)o[r];
            }
        }
    }
    __syncthreads();

    // ---- Phase D: projection (64x256)@(256x256)+bias -> out fp32 ----
    {
        const int ntile = wave;                      // 0..15
        v8f acc[WPB] = {};
        #pragma unroll
        for (int kc = 0; kc < KCH; ++kc) {
            const __bf16* bp = &wproj_pk[(size_t)((ntile * KCH + kc) * 32 + lane) * 16];
            if (kc < KCH - 1) __builtin_prefetch(bp + 512, 0, 0);
            v16bf bfr = frag_b(bp);
            #pragma unroll
            for (int mt = 0; mt < WPB; ++mt) {
                v16bf a = frag_a(&xyS[(mt * 16 + m16) * XY_STRIDE + kc * 32 + aoff]);
                acc[mt] = WMMA_BF16(a, bfr, acc[mt]);
            }
        }
        const int n = ntile * 16 + m16;
        const float bias = b_proj[n];
        #pragma unroll
        for (int mt = 0; mt < WPB; ++mt)
            #pragma unroll
            for (int r = 0; r < 8; ++r) {
                const int p = hi ? 8 + r : r;
                const int tp = p >> 2, sp = p & 3;
                const size_t row = (((size_t)b * TDIM) + (wt * 4 + tp)) * SDIM
                                 + ((wc0 + mt) * 4 + sp);
                out[row * D + n] = acc[mt][r] + bias;
            }
    }
}

extern "C" void kernel_launch(void* const* d_in, const int* in_sizes, int n_in,
                              void* d_out, int out_size, void* d_ws, size_t ws_size,
                              hipStream_t stream) {
    const float* x      = (const float*)d_in[0];
    const float* w_qkv  = (const float*)d_in[1];
    const float* w_proj = (const float*)d_in[2];
    const float* b_proj = (const float*)d_in[3];
    const float* rpb    = (const float*)d_in[4];
    float* out = (float*)d_out;

    __bf16* wqkv_pk  = (__bf16*)d_ws;                   // 196608 elems
    __bf16* wproj_pk = wqkv_pk + 48 * KCH * 32 * 16;    // 65536 elems

    pack_weights_kernel<<<(48 * KCH * 32 * 16 + 255) / 256, 256, 0, stream>>>(
        w_qkv, wqkv_pk, 48, QKV_N);
    pack_weights_kernel<<<(16 * KCH * 32 * 16 + 255) / 256, 256, 0, stream>>>(
        w_proj, wproj_pk, 16, D);

    window_attn_kernel<<<NBLK, 512, 0, stream>>>(
        x, wqkv_pk, wproj_pk, b_proj, rpb, out);
}
